// GCTErrorPredictor_81939386073700
// MI455X (gfx1250) — matrix operand order repaired and checked by reading, not compiled, not run on hardware
//
#include <hip/hip_runtime.h>
#include <hip/hip_bf16.h>

// ---------------------------------------------------------------------------
// GCT error predictor for MI455X (gfx1250, wave32, WMMA + TDM + async-LDS).
// N=4096 nodes, IN=16, HID=128, H=4 heads, D=32, L=2 layers, E=131072 edges.
// - All dense GEMMs + attention use v_wmma_f32_16x16x32_f16 (f32 accumulate);
//   weights are pre-transposed/converted to f16 so both A and B fragments
//   load as contiguous 16B chunks.
// - Attention stages K tiles via the Tensor Data Mover (tensor_load_to_lds
//   with pad_enable LDS row padding) and V tiles via async global->LDS,
//   synchronized with s_wait_tensorcnt / s_wait_asynccnt + split barriers.
// - Softmax row reductions use DPP8 xor butterflies + DPP16 row_ror:8
//   (pure VALU, no LDS round trips).
// - Dense [N,N] bias matrices live in workspace (128MB, resident in the
//   MI455X's 192MB L2), streamed with global_prefetch_b8.
// ---------------------------------------------------------------------------

#define NN    4096
#define INF_  16
#define HID   128
#define NHEAD 4
#define HD    32          // head dim
#define NLAY  2
#define NEDGE 131072
#define EPS   1e-5f

typedef __attribute__((ext_vector_type(16))) _Float16     v16h;
typedef __attribute__((ext_vector_type(8)))  _Float16     v8h;
typedef __attribute__((ext_vector_type(8)))  float        v8f;
typedef __attribute__((ext_vector_type(4)))  unsigned int u32x4;
typedef __attribute__((ext_vector_type(8)))  int          i32x8;
typedef __attribute__((ext_vector_type(4)))  int          i32x4;

static __device__ __forceinline__ v16h cat16(v8h a, v8h b) {
  return __builtin_shufflevector(a, b, 0,1,2,3,4,5,6,7,8,9,10,11,12,13,14,15);
}
static __device__ __forceinline__ v8f wmma_f16(v16h a, v16h b, v8f c) {
  // D = A(16x32 f16) * B(32x16 f16) + C(16x16 f32)
  return __builtin_amdgcn_wmma_f32_16x16x32_f16(false, a, false, b, (short)0, c,
                                                false, false);
}

// ---- DPP-based 16-lane row reductions (wave32: two independent rows) ------
constexpr int dpp8sel(int a,int b,int c,int d,int e,int f,int g,int h) {
  return a | (b<<3) | (c<<6) | (d<<9) | (e<<12) | (f<<15) | (g<<18) | (h<<21);
}
constexpr int DPP8_XOR1 = dpp8sel(1,0,3,2,5,4,7,6);
constexpr int DPP8_XOR2 = dpp8sel(2,3,0,1,6,7,4,5);
constexpr int DPP8_XOR4 = dpp8sel(4,5,6,7,0,1,2,3);

template <int SEL>
static __device__ __forceinline__ float dpp8_f(float v) {
  return __int_as_float(__builtin_amdgcn_mov_dpp8(__float_as_int(v), SEL));
}
static __device__ __forceinline__ float dpp_ror8_f(float v) {
  // DPP16 row_ror:8 == xor 8 within each 16-lane row
  return __int_as_float(__builtin_amdgcn_update_dpp(
      0, __float_as_int(v), 0x128, 0xf, 0xf, true));
}
static __device__ __forceinline__ float row_max16(float v) {
  v = fmaxf(v, dpp8_f<DPP8_XOR1>(v));
  v = fmaxf(v, dpp8_f<DPP8_XOR2>(v));
  v = fmaxf(v, dpp8_f<DPP8_XOR4>(v));
  v = fmaxf(v, dpp_ror8_f(v));
  return v;
}
static __device__ __forceinline__ float row_sum16(float v) {
  v += dpp8_f<DPP8_XOR1>(v);
  v += dpp8_f<DPP8_XOR2>(v);
  v += dpp8_f<DPP8_XOR4>(v);
  v += dpp_ror8_f(v);
  return v;
}

// ---- Async global->LDS copy of 8 bytes (ASYNCcnt path) --------------------
static __device__ __forceinline__ void async_copy_b64(unsigned ldsoff,
                                                      unsigned long long ga) {
  asm volatile("global_load_async_to_lds_b64 %0, %1, off"
               :: "v"(ldsoff), "v"(ga) : "memory");
}
static __device__ __forceinline__ void wait_asynccnt0() {
  asm volatile("s_wait_asynccnt 0" ::: "memory");
}

// ---- TDM: [32 x 32] f16 tile from a [4096, 128] f16 tensor into LDS -------
// pad_enable inserts 4 DWORDs per 16 DWORDs -> LDS row pitch 80B.
static __device__ __forceinline__ void tdm_load_k_tile(unsigned lds_addr,
                                                       unsigned long long ga) {
  u32x4 g0 = { 1u, lds_addr, (unsigned)ga,
               (unsigned)((ga >> 32) & 0x01FFFFFFull) | 0x80000000u };
  i32x8 g1 = { (int)(0x00010000u | (1u << 20) | (3u << 22) | (3u << 25)),
               (int)(128u << 16),            // tensor_dim0 lo16 @ bits[63:48]
               (int)(4096u << 16),           // dim0 hi16 | tensor_dim1 lo16
               (int)(32u << 16),             // dim1 hi16 | tile_dim0
               32,                           // tile_dim1 | tile_dim2=0
               128,                          // tensor_dim0_stride lo32
               0, 0 };
  i32x4 g2 = {}, g3 = {};
#if defined(__clang_major__) && __clang_major__ >= 23
  i32x8 g4 = {};
  __builtin_amdgcn_tensor_load_to_lds(g0, g1, g2, g3, g4, 0);
#else
  __builtin_amdgcn_tensor_load_to_lds(g0, g1, g2, g3, 0);
#endif
}

// ---------------------------------------------------------------------------
// Workspace zeroing (W_mat / C_mat) -- float4 grid-stride
// ---------------------------------------------------------------------------
__global__ void k_zero(float4* __restrict__ p, size_t n4) {
  size_t i = (size_t)blockIdx.x * blockDim.x + threadIdx.x;
  size_t stride = (size_t)gridDim.x * blockDim.x;
  float4 z = make_float4(0.f, 0.f, 0.f, 0.f);
  for (; i < n4; i += stride) p[i] = z;
}

// ---------------------------------------------------------------------------
// Sparse edge scatter -> dense W_mat, C_mat  [N,N]
// ---------------------------------------------------------------------------
__global__ void k_scatter(const int* __restrict__ eidx,
                          const float* __restrict__ ew,
                          const float* __restrict__ ec,
                          float* __restrict__ Wm, float* __restrict__ Cm) {
  int e = blockIdx.x * blockDim.x + threadIdx.x;
  if (e >= NEDGE) return;
  int s = eidx[e];
  int d = eidx[NEDGE + e];
  size_t off = (size_t)s * NN + d;
  Wm[off] = ew[e];
  Cm[off] = ec[e];
}

// ---------------------------------------------------------------------------
// Weight prep: transpose + convert f32 [K, Nout] -> f16 [Nout, K]
// (one-time; makes GEMM B-fragments contiguous 16B loads)
// ---------------------------------------------------------------------------
__global__ void k_prep_w(const float* __restrict__ in, _Float16* __restrict__ out,
                         int K, int Nout) {
  int i = blockIdx.x * blockDim.x + threadIdx.x;
  if (i >= K * Nout) return;
  int r = i / Nout, c = i % Nout;
  out[(size_t)c * K + r] = (_Float16)in[i];
}

// ---------------------------------------------------------------------------
// node_proj front half: t = LN(x @ w1 + b1); a16 = relu(t) in f16
// ---------------------------------------------------------------------------
__global__ void k_node_proj(const float* __restrict__ x,
                            const float* __restrict__ w1,
                            const float* __restrict__ b1,
                            const float* __restrict__ lng,
                            const float* __restrict__ lnb,
                            _Float16* __restrict__ a16) {
  int r = blockIdx.x * blockDim.x + threadIdx.x;
  if (r >= NN) return;
  float xr[INF_];
#pragma unroll
  for (int i = 0; i < INF_; ++i) xr[i] = x[r * INF_ + i];
  float t[HID];
  float s1 = 0.f, s2 = 0.f;
#pragma unroll 4
  for (int j = 0; j < HID; ++j) {
    float acc = b1[j];
#pragma unroll
    for (int i = 0; i < INF_; ++i) acc = fmaf(xr[i], w1[i * HID + j], acc);
    t[j] = acc;
    s1 += acc;
    s2 += acc * acc;
  }
  float mean = s1 * (1.f / HID);
  float var  = s2 * (1.f / HID) - mean * mean;
  float rstd = rsqrtf(var + EPS);
#pragma unroll 4
  for (int j = 0; j < HID; ++j) {
    float v = (t[j] - mean) * rstd * lng[j] + lnb[j];
    v = fmaxf(v, 0.f);
    a16[(size_t)r * HID + j] = (_Float16)v;
  }
}

// ---------------------------------------------------------------------------
// Generic WMMA GEMM:  out = act(A16[M,K] @ Bt16[Nout,K]^T + bias)
// A f16 row-major; Bt f16 transposed (weight-stationary prepped layout).
// One wave per 16x16 output tile, grid-stride over tiles.
// A and B fragments both use contiguous k order (lane half h holds k = 16h + s).
// ---------------------------------------------------------------------------
template <int K, int NOUT, bool RELU>
__global__ void k_wmma_gemm(const _Float16* __restrict__ A,
                            const _Float16* __restrict__ Bt,
                            const float* __restrict__ bias,
                            float* __restrict__ outF,
                            _Float16* __restrict__ outH, int M) {
  const int lane = threadIdx.x & 31;
  const int lp   = lane & 15;
  const bool hi  = lane >= 16;
  const int tilesN = NOUT / 16;
  const int tiles  = (M / 16) * tilesN;
  int wave  = (blockIdx.x * blockDim.x + threadIdx.x) >> 5;
  int nwave = (gridDim.x * blockDim.x) >> 5;

  for (int t = wave; t < tiles; t += nwave) {
    const int tm = t / tilesN, tn = t % tilesN;
    const int row = tm * 16 + lp;
    const int col = tn * 16 + lp;
    v8f acc = {};
    const int aoff = hi ? 16 : 0;
    const int boff = hi ? 16 : 0;
#pragma unroll 2
    for (int kk = 0; kk < K; kk += 32) {
      const _Float16* ap = A + (size_t)row * K + kk + aoff;
      v16h af = cat16(*reinterpret_cast<const v8h*>(ap),
                      *reinterpret_cast<const v8h*>(ap + 8));
      const _Float16* bp = Bt + (size_t)col * K + kk + boff;
      v16h bf = cat16(*reinterpret_cast<const v8h*>(bp),
                      *reinterpret_cast<const v8h*>(bp + 8));
      acc = wmma_f16(af, bf, acc);
    }
    const float bb = bias ? bias[col] : 0.f;
#pragma unroll
    for (int i = 0; i < 8; ++i) {
      int r = tm * 16 + i + (hi ? 8 : 0);
      float v = acc[i] + bb;
      if (RELU) v = fmaxf(v, 0.f);
      if (outF) outF[(size_t)r * NOUT + col] = v;
      if (outH) outH[(size_t)r * NOUT + col] = (_Float16)v;
    }
  }
}

// ---------------------------------------------------------------------------
// Flash attention with dense edge bias; one head per blockIdx.y.
// Block = 256 threads (8 waves); wave w owns query rows [128*bx + 16w, +16).
// Per 32-key block: TDM stages the K tile (LDS row padding), async
// global->LDS stages the V tile; 2 WMMA (Q K^T) + DPP online softmax + LDS
// transpose of P + 2 WMMA (P V).  Bias tiles stream from the dense W/C
// matrices (L2-resident), prefetched one block ahead.
// ---------------------------------------------------------------------------
#define KV_PITCH 40   // halves per LDS tile row (32 data + 8 pad = 80B)

__global__ void k_attn(const _Float16* __restrict__ q16,
                       const _Float16* __restrict__ k16,
                       const _Float16* __restrict__ v16p,
                       const float* __restrict__ Wm,
                       const float* __restrict__ Cm,
                       const float* __restrict__ bwp,
                       const float* __restrict__ bcp,
                       _Float16* __restrict__ attn16) {
  __shared__ _Float16 sK[32 * KV_PITCH];   // K tile, key-major, padded pitch
  __shared__ _Float16 sV[32 * KV_PITCH];   // V tile, key-major, padded pitch
  __shared__ _Float16 sP[8][16][32];       // per-wave P transpose buffer

  const int tid  = threadIdx.x;
  const int w    = tid >> 5;
  const int lane = tid & 31;
  const int lp   = lane & 15;
  const bool hi  = lane >= 16;
  const int head = blockIdx.y;
  const int qbase = blockIdx.x * 128 + w * 16;
  const float scale = 0.1767766952966369f;  // 1/sqrt(32)
  const float bw = bwp[head], bc = bcp[head];

  const unsigned sK_off = (unsigned)(uintptr_t)(const void*)sK;
  const unsigned sV_off = (unsigned)(uintptr_t)(const void*)sV;
  const int cpKey = tid >> 3;        // staging: thread -> (key, 8B chunk)
  const int cpCh  = tid & 7;

  // Q fragment (A layout), loaded once from global.
  const _Float16* qrow = q16 + (size_t)(qbase + lp) * HID + head * HD + (hi ? 16 : 0);
  v16h qa = cat16(*reinterpret_cast<const v8h*>(qrow),
                  *reinterpret_cast<const v8h*>(qrow + 8));

  v8f o0 = {}, o1 = {};
  float m[8], l[8];
#pragma unroll
  for (int i = 0; i < 8; ++i) { m[i] = -3.0e38f; l[i] = 0.f; }

  _Float16* pp = &sP[w][0][0];

  for (int kb = 0; kb < NN; kb += 32) {
    // ---- stage K tile via TDM (wave 0), V tile via async LDS copies ----
    if (w == 0) {
      tdm_load_k_tile(sK_off,
          (unsigned long long)(uintptr_t)(k16 + (size_t)kb * HID + head * HD));
    }
    async_copy_b64(sV_off + (unsigned)(cpKey * (KV_PITCH * 2) + cpCh * 8),
        (unsigned long long)(uintptr_t)(v16p + (size_t)(kb + cpKey) * HID +
                                        head * HD + cpCh * 4));
    __builtin_amdgcn_s_wait_tensorcnt(0);
    wait_asynccnt0();
    __syncthreads();

    // ---- K fragments (B layout) from LDS: halves = dims, col = key ----
    const _Float16* kp0 = sK + (size_t)lp * KV_PITCH + (hi ? 16 : 0);
    const _Float16* kp1 = sK + (size_t)(16 + lp) * KV_PITCH + (hi ? 16 : 0);
    v16h kf0 = cat16(*reinterpret_cast<const v8h*>(kp0),
                     *reinterpret_cast<const v8h*>(kp0 + 8));
    v16h kf1 = cat16(*reinterpret_cast<const v8h*>(kp1),
                     *reinterpret_cast<const v8h*>(kp1 + 8));

    // ---- V fragments (B layout) from LDS: halves = keys, col = dim ----
    v16h vf0, vf1;
#pragma unroll
    for (int i = 0; i < 16; ++i) {
      const _Float16* vp = sV + (size_t)(i + (hi ? 16 : 0)) * KV_PITCH;
      vf0[i] = vp[lp];
      vf1[i] = vp[16 + lp];
    }

    v8f s0 = {}, s1 = {};
    s0 = wmma_f16(qa, kf0, s0);
    s1 = wmma_f16(qa, kf1, s1);

    // ---- scale + dense edge bias (L2-resident), prefetch next block ----
    if (kb + 32 < NN) {
      __builtin_prefetch(Wm + (size_t)(qbase + lp) * NN + kb + 32, 0, 0);
      __builtin_prefetch(Cm + (size_t)(qbase + lp) * NN + kb + 32, 0, 0);
    }
#pragma unroll
    for (int i = 0; i < 8; ++i) {
      int r = qbase + i + (hi ? 8 : 0);
      const float* wr = Wm + (size_t)r * NN + kb;
      const float* cr = Cm + (size_t)r * NN + kb;
      s0[i] = fmaf(s0[i], scale, fmaf(bw, wr[lp],      bc * cr[lp]));
      s1[i] = fmaf(s1[i], scale, fmaf(bw, wr[16 + lp], bc * cr[16 + lp]));
    }

    // ---- online softmax (DPP butterflies, no LDS) ----
#pragma unroll
    for (int i = 0; i < 8; ++i) {
      float v = row_max16(fmaxf(s0[i], s1[i]));
      float mn = fmaxf(m[i], v);
      float alpha = __expf(m[i] - mn);
      float p0 = __expf(s0[i] - mn);
      float p1 = __expf(s1[i] - mn);
      s0[i] = p0; s1[i] = p1;
      float ss = row_sum16(p0 + p1);
      l[i] = l[i] * alpha + ss;
      m[i] = mn;
      o0[i] *= alpha;
      o1[i] *= alpha;
    }

    // ---- transpose P (C layout -> A layout) through LDS ----
#pragma unroll
    for (int i = 0; i < 8; ++i) {
      int rr = i + (hi ? 8 : 0);
      pp[rr * 32 + lp]      = (_Float16)s0[i];
      pp[rr * 32 + 16 + lp] = (_Float16)s1[i];
    }
    // Mid barrier: publishes P AND retires all K/V tile reads, so next
    // iteration's TDM/async staging cannot race them.
    __syncthreads();
    v16h pa = cat16(*reinterpret_cast<const v8h*>(pp + lp * 32 + (hi ? 16 : 0)),
                    *reinterpret_cast<const v8h*>(pp + lp * 32 + 8 + (hi ? 16 : 0)));

    o0 = wmma_f16(pa, vf0, o0);
    o1 = wmma_f16(pa, vf1, o1);
  }

  // ---- finalize: divide by row sums (fast rcp), write f16 ----
#pragma unroll
  for (int i = 0; i < 8; ++i) {
    int r = qbase + i + (hi ? 8 : 0);
    float inv = __builtin_amdgcn_rcpf(l[i]);
    size_t base = (size_t)r * HID + head * HD;
    attn16[base + lp]      = (_Float16)(o0[i] * inv);
    attn16[base + 16 + lp] = (_Float16)(o1[i] * inv);
  }
}

// ---------------------------------------------------------------------------
// Residual + LayerNorm: h = LN(h + t) * g + b  -> hF (f32) and h16 (f16)
// ---------------------------------------------------------------------------
__global__ void k_add_ln(float* __restrict__ hF,
                         const float* __restrict__ t,
                         const float* __restrict__ g,
                         const float* __restrict__ b,
                         _Float16* __restrict__ h16) {
  int r = blockIdx.x * blockDim.x + threadIdx.x;
  if (r >= NN) return;
  float v[HID];
  float s1 = 0.f, s2 = 0.f;
#pragma unroll 4
  for (int j = 0; j < HID; ++j) {
    float x = hF[(size_t)r * HID + j] + t[(size_t)r * HID + j];
    v[j] = x;
    s1 += x;
    s2 += x * x;
  }
  float mean = s1 * (1.f / HID);
  float var  = s2 * (1.f / HID) - mean * mean;
  float rstd = rsqrtf(var + EPS);
#pragma unroll 4
  for (int j = 0; j < HID; ++j) {
    float y = (v[j] - mean) * rstd * g[j] + b[j];
    hF[(size_t)r * HID + j]  = y;
    h16[(size_t)r * HID + j] = (_Float16)y;
  }
}

// ---------------------------------------------------------------------------
// Output heads: root/multi logits.
// ---------------------------------------------------------------------------
__global__ void k_heads(const float* __restrict__ h,
                        const float* __restrict__ rw, const float* __restrict__ rb,
                        const float* __restrict__ mw, const float* __restrict__ mb,
                        float* __restrict__ out) {
  int r = blockIdx.x * blockDim.x + threadIdx.x;
  if (r >= NN) return;
  float a = 0.f, c = 0.f;
#pragma unroll 4
  for (int j = 0; j < HID; ++j) {
    float x = h[(size_t)r * HID + j];
    a = fmaf(x, rw[j], a);
    c = fmaf(x, mw[j], c);
  }
  out[r]      = a + rb[0];
  out[NN + r] = c + mb[0];
}

// ---------------------------------------------------------------------------
// Host-side orchestration
// ---------------------------------------------------------------------------
extern "C" void kernel_launch(void* const* d_in, const int* in_sizes, int n_in,
                              void* d_out, int out_size, void* d_ws, size_t ws_size,
                              hipStream_t stream) {
  const float* x      = (const float*)d_in[0];
  const int*   eidx   = (const int*)d_in[1];
  const float* ew     = (const float*)d_in[2];
  const float* ec     = (const float*)d_in[3];
  const float* np_w1  = (const float*)d_in[4];
  const float* np_b1  = (const float*)d_in[5];
  const float* np_lng = (const float*)d_in[6];
  const float* np_lnb = (const float*)d_in[7];
  const float* np_w2  = (const float*)d_in[8];
  const float* np_b2  = (const float*)d_in[9];
  const float* Wq     = (const float*)d_in[10];
  const float* Wk     = (const float*)d_in[11];
  const float* Wv     = (const float*)d_in[12];
  const float* Wo     = (const float*)d_in[13];
  const float* bo     = (const float*)d_in[14];
  const float* b_w    = (const float*)d_in[15];
  const float* b_c    = (const float*)d_in[16];
  const float* ln1_g  = (const float*)d_in[17];
  const float* ln1_b  = (const float*)d_in[18];
  const float* ln2_g  = (const float*)d_in[19];
  const float* ln2_b  = (const float*)d_in[20];
  const float* ff_w1  = (const float*)d_in[21];
  const float* ff_b1  = (const float*)d_in[22];
  const float* ff_w2  = (const float*)d_in[23];
  const float* ff_b2  = (const float*)d_in[24];
  const float* root_w = (const float*)d_in[25];
  const float* root_b = (const float*)d_in[26];
  const float* multi_w= (const float*)d_in[27];
  const float* multi_b= (const float*)d_in[28];
  float* out = (float*)d_out;

  // Workspace carve-up
  char* p = (char*)d_ws;
  const size_t MAT = (size_t)NN * NN * sizeof(float);     // 64 MB
  float*    Wm    = (float*)p;              p += MAT;
  float*    Cm    = (float*)p;              p += MAT;
  float*    hF    = (float*)p;              p += (size_t)NN * HID * 4;
  float*    tmpF  = (float*)p;              p += (size_t)NN * HID * 4;
  _Float16* h16   = (_Float16*)p;           p += (size_t)NN * HID * 2;
  _Float16* a16   = (_Float16*)p;           p += (size_t)NN * HID * 2;
  _Float16* q16   = (_Float16*)p;           p += (size_t)NN * HID * 2;
  _Float16* k16   = (_Float16*)p;           p += (size_t)NN * HID * 2;
  _Float16* v16w  = (_Float16*)p;           p += (size_t)NN * HID * 2;
  _Float16* at16  = (_Float16*)p;           p += (size_t)NN * HID * 2;
  _Float16* ff16  = (_Float16*)p;           p += (size_t)NN * 2 * HID * 2;
  // Pre-transposed f16 weights (B operands)
  _Float16* w2t   = (_Float16*)p;           p += (size_t)HID * HID * 2;
  _Float16* qt[NLAY]; _Float16* kt[NLAY]; _Float16* vt[NLAY]; _Float16* ot[NLAY];
  _Float16* f1t[NLAY]; _Float16* f2t[NLAY];
  for (int l = 0; l < NLAY; ++l) {
    qt[l]  = (_Float16*)p; p += (size_t)HID * HID * 2;
    kt[l]  = (_Float16*)p; p += (size_t)HID * HID * 2;
    vt[l]  = (_Float16*)p; p += (size_t)HID * HID * 2;
    ot[l]  = (_Float16*)p; p += (size_t)HID * HID * 2;
    f1t[l] = (_Float16*)p; p += (size_t)HID * 2 * HID * 2;
    f2t[l] = (_Float16*)p; p += (size_t)2 * HID * HID * 2;
  }

  // 1) zero + scatter dense bias matrices
  k_zero<<<4096, 256, 0, stream>>>((float4*)Wm, (2 * MAT) / 16);
  k_scatter<<<NEDGE / 256, 256, 0, stream>>>(eidx, ew, ec, Wm, Cm);

  // 1b) weight prep: transpose + f16-convert all GEMM B operands
  const int E1 = HID * HID, E2 = HID * 2 * HID;
  k_prep_w<<<(E1 + 255) / 256, 256, 0, stream>>>(np_w2, w2t, HID, HID);
  for (int l = 0; l < NLAY; ++l) {
    const size_t wo = (size_t)l * HID * HID;
    k_prep_w<<<(E1 + 255) / 256, 256, 0, stream>>>(Wq + wo, qt[l], HID, HID);
    k_prep_w<<<(E1 + 255) / 256, 256, 0, stream>>>(Wk + wo, kt[l], HID, HID);
    k_prep_w<<<(E1 + 255) / 256, 256, 0, stream>>>(Wv + wo, vt[l], HID, HID);
    k_prep_w<<<(E1 + 255) / 256, 256, 0, stream>>>(Wo + wo, ot[l], HID, HID);
    k_prep_w<<<(E2 + 255) / 256, 256, 0, stream>>>(ff_w1 + (size_t)l * E2, f1t[l], HID, 2 * HID);
    k_prep_w<<<(E2 + 255) / 256, 256, 0, stream>>>(ff_w2 + (size_t)l * E2, f2t[l], 2 * HID, HID);
  }

  // 2) node_proj: LN(x@w1+b1) -> relu (f16), then WMMA x np_w2
  k_node_proj<<<NN / 256, 256, 0, stream>>>(x, np_w1, np_b1, np_lng, np_lnb, a16);
  k_wmma_gemm<HID, HID, false><<<256, 256, 0, stream>>>(a16, w2t, np_b2, hF, h16, NN);

  // 3) transformer layers
  for (int l = 0; l < NLAY; ++l) {
    k_wmma_gemm<HID, HID, false><<<256, 256, 0, stream>>>(h16, qt[l], nullptr, nullptr, q16, NN);
    k_wmma_gemm<HID, HID, false><<<256, 256, 0, stream>>>(h16, kt[l], nullptr, nullptr, k16, NN);
    k_wmma_gemm<HID, HID, false><<<256, 256, 0, stream>>>(h16, vt[l], nullptr, nullptr, v16w, NN);

    dim3 ag(NN / 128, NHEAD);
    k_attn<<<ag, 256, 0, stream>>>(q16, k16, v16w, Wm, Cm,
                                   b_w + l * NHEAD, b_c + l * NHEAD, at16);

    k_wmma_gemm<HID, HID, false><<<256, 256, 0, stream>>>(at16, ot[l], bo + l * HID,
                                                          tmpF, nullptr, NN);
    k_add_ln<<<NN / 256, 256, 0, stream>>>(hF, tmpF, ln1_g + l * HID, ln1_b + l * HID, h16);

    k_wmma_gemm<HID, 2 * HID, true><<<512, 256, 0, stream>>>(
        h16, f1t[l], ff_b1 + l * 2 * HID, nullptr, ff16, NN);
    k_wmma_gemm<2 * HID, HID, false><<<256, 256, 0, stream>>>(
        ff16, f2t[l], ff_b2 + l * HID, tmpF, nullptr, NN);
    k_add_ln<<<NN / 256, 256, 0, stream>>>(hF, tmpF, ln2_g + l * HID, ln2_b + l * HID, h16);
  }

  // 4) output heads
  k_heads<<<NN / 256, 256, 0, stream>>>(hF, root_w, root_b, multi_w, multi_b, out);
}
